// WindowAttention_45844480917815
// MI455X (gfx1250) — compile-verified, matrix operand order checked by reading, not run on hardware
//
#include <hip/hip_runtime.h>

// ---------------------------------------------------------------------------
// MI455X (gfx1250) fused window-attention.
// Precision: bf16 WMMA (v_wmma_f32_16x16x32_bf16) with fp32 accumulation.
// One workgroup (256 threads = 8 wave32) per window; everything fused in LDS.
// Weights are prepacked once per launch into WMMA-B-fragment order in d_ws.
// ---------------------------------------------------------------------------

typedef __attribute__((ext_vector_type(16))) __bf16 v16bf;
typedef __attribute__((ext_vector_type(8)))  float  v8f;
typedef __attribute__((ext_vector_type(4)))  unsigned int u32x4;

#define NTOK 49        // tokens per window
#define MP   64        // padded token count (4 x 16)
#define DIMC 192
#define C2C  384       // 2*DIM
#define QKVC 1152      // 6*DIM
#define NH   6
#define DH   64
#define NWIN 64        // mask windows
#define SCALEQ 0.17677669529663687f   // (192/6)^-0.5

// LDS strides (elements). All multiples of 8 (=> 16B-aligned b128 loads),
// row byte strides not multiples of 256B (bank spread).
#define XY_STRIDE 392  // 64 x 392 bf16 (xy staging, padded rows zeroed)
#define QK_STRIDE 72   // 64 x 72 bf16 (Q, K, V^T, P)
#define S_STRIDE  68   // 64 x 68 f32  (scores / softmax workspace)
#define AO_STRIDE 392  // 64 x 392 bf16 (concat head outputs)

union F16x16 { v16bf bf; u32x4 q[2]; };

__device__ __forceinline__ unsigned short f2bf(float x) {
  unsigned int u = __float_as_uint(x);
  unsigned int r = (u + 0x7FFFu + ((u >> 16) & 1u)) >> 16;  // round-nearest-even
  return (unsigned short)r;
}

// A/B fragment from a row-major bf16 matrix in LDS.
// lane 0-15 : row = row0+lane,    K = {k0..k0+7,  k0+16..k0+23}
// lane16-31 : row = row0+lane-16, K = {k0+8..+15, k0+24..k0+31}
// => two contiguous 16B LDS loads per fragment (ds_load_b128).
__device__ __forceinline__ v16bf load_frag_row(const unsigned short* base,
                                               int row0, int k0, int stride, int lane) {
  const int r  = row0 + (lane & 15);
  const int kb = k0 + ((lane & 16) ? 8 : 0);
  const unsigned short* p = base + r * stride + kb;
  F16x16 f;
  f.q[0] = *(const u32x4*)(p);        // K = kb .. kb+7
  f.q[1] = *(const u32x4*)(p + 16);   // K = kb+16 .. kb+23
  return f.bf;
}

// B fragment from the prepacked weight buffer: 32 contiguous bytes per lane,
// lane-major => perfectly coalesced pair of global b128 loads.
__device__ __forceinline__ v16bf load_frag_packed(const unsigned short* wpk,
                                                  int kt, int ntile, int ntiles, int lane) {
  const unsigned short* p = wpk + ((size_t)((kt * ntiles + ntile) * 32 + lane)) * 16;
  F16x16 f;
  f.q[0] = *(const u32x4*)(p);
  f.q[1] = *(const u32x4*)(p + 8);
  return f.bf;
}

__device__ __forceinline__ v8f wmma_bf16(v16bf a, v16bf b, v8f c) {
  return __builtin_amdgcn_wmma_f32_16x16x32_bf16(false, a, false, b, (short)0, c,
                                                 false, false);
}

// ---------------------------------------------------------------------------
// Prepack kernel: W [Nfull][Kfull] fp32 row-major -> bf16 B-fragment order for
// B = W^T (K x N). Element order: [kt][nt][lane][i] with
//   n = nt*16 + lane%16 ; k = kt*32 + kmap(lane, i).
// ---------------------------------------------------------------------------
__global__ void prepack_wmma_b(const float* __restrict__ W,
                               unsigned short* __restrict__ out,
                               int Nfull, int Kfull) {
  int idx = blockIdx.x * blockDim.x + threadIdx.x;
  int total = (Kfull >> 5) * (Nfull >> 4) * 32 * 16;
  if (idx >= total) return;
  int i    = idx & 15;
  int lane = (idx >> 4) & 31;
  int blk  = idx >> 9;
  int ntiles = Nfull >> 4;
  int nt = blk % ntiles;
  int kt = blk / ntiles;
  int n  = nt * 16 + (lane & 15);
  int kb = (lane & 16) ? 8 : 0;
  int k  = kt * 32 + ((i < 8) ? (kb + i) : (16 + kb + (i - 8)));
  out[idx] = f2bf(W[(size_t)n * Kfull + k]);
}

// ---------------------------------------------------------------------------
// Fused window attention: one block per window.
// ---------------------------------------------------------------------------
__global__ __launch_bounds__(256)
void winattn_kernel(const float* __restrict__ x, const float* __restrict__ y,
                    const float* __restrict__ mask,
                    const float* __restrict__ b_qkv, const float* __restrict__ b_proj,
                    const float* __restrict__ bias_table,
                    const unsigned short* __restrict__ wqkv_pk,
                    const unsigned short* __restrict__ wproj_pk,
                    float* __restrict__ out) {
  extern __shared__ char smem[];
  unsigned short* sXY = (unsigned short*)smem;            // 64*392 bf16
  unsigned short* sQ  = sXY + MP * XY_STRIDE;             // 64*72
  unsigned short* sK  = sQ  + MP * QK_STRIDE;             // 64*72
  unsigned short* sVt = sK  + MP * QK_STRIDE;             // 64*72  (V transposed: [feat][tok])
  float*          sS  = (float*)(sVt + MP * QK_STRIDE);   // 64*68 f32
  unsigned short* sP  = (unsigned short*)(sS + MP * S_STRIDE); // 64*72
  unsigned short* sAO = sP  + MP * QK_STRIDE;             // 64*392

  const int tid  = threadIdx.x;
  const int lane = tid & 31;
  const int wave = tid >> 5;
  const int b    = blockIdx.x;
  const int mwin = b & (NWIN - 1);

  // ---- stage xy = concat(x, y) into LDS (bf16), zero-pad rows 49..63 ----
  for (int idx = tid; idx < MP * C2C; idx += 256) {
    int r = idx / C2C, c = idx % C2C;
    float v = 0.0f;
    if (r < NTOK)
      v = (c < DIMC) ? x[((size_t)b * NTOK + r) * DIMC + c]
                     : y[((size_t)b * NTOK + r) * DIMC + (c - DIMC)];
    sXY[r * XY_STRIDE + c] = f2bf(v);
  }
  __syncthreads();

  const int colL  = lane & 15;
  const int rHi   = (lane & 16) ? 8 : 0;

  for (int h = 0; h < NH; ++h) {
    // ---- QKV projection for this head: 3 matrices x (4x4) 16x16 tiles ----
    for (int s = 0; s < 6; ++s) {
      int t = wave * 6 + s;              // 0..47
      int mat   = t >> 4;                // 0=Q 1=K 2=V
      int rem   = t & 15;
      int mtile = rem >> 2;
      int ntile = rem & 3;
      int ntg   = mat * 24 + h * 4 + ntile;   // global n-tile in 1152-wide QKV
      v8f acc = {};
      for (int kt = 0; kt < 12; ++kt) {
        v16bf a  = load_frag_row(sXY, mtile * 16, kt * 32, XY_STRIDE, lane);
        v16bf bb = load_frag_packed(wqkv_pk, kt, ntg, QKVC / 16, lane);
        acc = wmma_bf16(a, bb, acc);
      }
      int col  = ntile * 16 + colL;
      float bv = b_qkv[mat * C2C + h * DH + col];
      int rb   = mtile * 16 + rHi;
      for (int j = 0; j < 8; ++j) {
        int row = rb + j;
        float v = acc[j] + bv;
        if (mat == 0)      sQ [row * QK_STRIDE + col] = f2bf(v * SCALEQ);
        else if (mat == 1) sK [row * QK_STRIDE + col] = f2bf(v);
        else               sVt[col * QK_STRIDE + row] = f2bf(v);  // transpose V
      }
    }
    __syncthreads();

    // ---- S = Q @ K^T  (+ relative-position bias + window mask) ----
    for (int s = 0; s < 2; ++s) {
      int t = wave * 2 + s;              // 0..15
      int mtile = t >> 2, ntile = t & 3;
      v8f acc = {};
      for (int kt = 0; kt < 2; ++kt) {
        v16bf a  = load_frag_row(sQ, mtile * 16, kt * 32, QK_STRIDE, lane);
        v16bf bb = load_frag_row(sK, ntile * 16, kt * 32, QK_STRIDE, lane);
        acc = wmma_bf16(a, bb, acc);
      }
      int cj = ntile * 16 + colL;
      int rb = mtile * 16 + rHi;
      for (int j = 0; j < 8; ++j) {
        int row = rb + j;
        float v;
        if (row < NTOK && cj < NTOK) {
          // closed-form relative position index (7x7 window, 13x13 table)
          int ridx = (row / 7 - cj / 7 + 6) * 13 + (row % 7 - cj % 7 + 6);
          v = acc[j]
            + bias_table[ridx * NH + h]
            + mask[((size_t)mwin * NTOK + row) * NTOK + cj];
        } else {
          v = -1e30f;                    // padded keys -> zero probability
        }
        sS[row * S_STRIDE + cj] = v;
      }
    }
    __syncthreads();

    // ---- row softmax over 64 (padded) columns ----
    if (tid < MP) {
      float* srow = sS + tid * S_STRIDE;
      float m = -1e30f;
      for (int j = 0; j < MP; ++j) m = fmaxf(m, srow[j]);
      float sum = 0.0f;
      for (int j = 0; j < MP; ++j) { float e = __expf(srow[j] - m); srow[j] = e; sum += e; }
      float inv = 1.0f / sum;
      unsigned short* prow = sP + tid * QK_STRIDE;
      for (int j = 0; j < MP; ++j) prow[j] = f2bf(srow[j] * inv);
    }
    __syncthreads();

    // ---- O = P @ V  (V stored transposed -> contiguous B fragments) ----
    for (int s = 0; s < 2; ++s) {
      int t = wave * 2 + s;
      int mtile = t >> 2, ntile = t & 3;
      v8f acc = {};
      for (int kt = 0; kt < 2; ++kt) {
        v16bf a  = load_frag_row(sP,  mtile * 16, kt * 32, QK_STRIDE, lane);
        v16bf bb = load_frag_row(sVt, ntile * 16, kt * 32, QK_STRIDE, lane);
        acc = wmma_bf16(a, bb, acc);
      }
      int cj = ntile * 16 + colL;
      int rb = mtile * 16 + rHi;
      for (int j = 0; j < 8; ++j)
        sAO[(rb + j) * AO_STRIDE + h * DH + cj] = f2bf(acc[j]);
    }
    __syncthreads();
  }

  // ---- output projection: [64,384] @ w_proj^T -> [49,192] ----
  for (int s = 0; s < 6; ++s) {
    int t = wave * 6 + s;                // 0..47 : 4 mtiles x 12 ntiles
    int mtile = t / 12, ntile = t % 12;
    v8f acc = {};
    for (int kt = 0; kt < 12; ++kt) {
      v16bf a  = load_frag_row(sAO, mtile * 16, kt * 32, AO_STRIDE, lane);
      v16bf bb = load_frag_packed(wproj_pk, kt, ntile, DIMC / 16, lane);
      acc = wmma_bf16(a, bb, acc);
    }
    int col  = ntile * 16 + colL;
    float bv = b_proj[col];
    int rb   = mtile * 16 + rHi;
    for (int j = 0; j < 8; ++j) {
      int row = rb + j;
      if (row < NTOK)
        out[((size_t)b * NTOK + row) * DIMC + col] = acc[j] + bv;
    }
  }
}

// ---------------------------------------------------------------------------
extern "C" void kernel_launch(void* const* d_in, const int* in_sizes, int n_in,
                              void* d_out, int out_size, void* d_ws, size_t ws_size,
                              hipStream_t stream) {
  (void)in_sizes; (void)n_in; (void)out_size; (void)ws_size;
  const float* x          = (const float*)d_in[0];
  const float* y          = (const float*)d_in[1];
  const float* mask       = (const float*)d_in[2];
  const float* w_qkv      = (const float*)d_in[3];
  const float* b_qkv      = (const float*)d_in[4];
  const float* w_proj     = (const float*)d_in[5];
  const float* b_proj     = (const float*)d_in[6];
  const float* bias_table = (const float*)d_in[7];

  // workspace: prepacked bf16 weights (~1.0 MB total)
  unsigned short* wq_pk = (unsigned short*)d_ws;              // 1152*384 bf16
  unsigned short* wp_pk = wq_pk + (size_t)QKVC * C2C;         //  192*384 bf16

  int nq = (C2C / 32) * (QKVC / 16) * 32 * 16;  // 442368
  int np = (C2C / 32) * (DIMC / 16) * 32 * 16;  //  73728
  prepack_wmma_b<<<(nq + 255) / 256, 256, 0, stream>>>(w_qkv, wq_pk, QKVC, C2C);
  prepack_wmma_b<<<(np + 255) / 256, 256, 0, stream>>>(w_proj, wp_pk, DIMC, C2C);

  // dynamic LDS: xy(50176) + Q/K/Vt(3*9216) + S(17408) + P(9216) + AO(50176)
  size_t shmem = (size_t)MP * XY_STRIDE * 2 + 3 * (size_t)MP * QK_STRIDE * 2
               + (size_t)MP * S_STRIDE * 4 + (size_t)MP * QK_STRIDE * 2
               + (size_t)MP * AO_STRIDE * 2;   // = 154624 B -> 2 blocks/WGP

  winattn_kernel<<<2048, 256, shmem, stream>>>(x, y, mask, b_qkv, b_proj,
                                               bias_table, wq_pk, wp_pk,
                                               (float*)d_out);
}
